// fftNotesLayer_37958920962278
// MI455X (gfx1250) — matrix operand order repaired and checked by reading, not compiled, not run on hardware
//
#include <hip/hip_runtime.h>
#include <hip/hip_bf16.h>

typedef __attribute__((ext_vector_type(16))) _Float16 v16h;
typedef __attribute__((ext_vector_type(8)))  float    v8f;

#define WIN   442
#define HOP   4
#define NBINS 221
#define NKS   14           // K steps of 32 (442 padded to 448)
#define NNT   14           // N tiles of 16 (224 >= 221)
#define THR2  (0.01f * 0.01f)
#define RELC  0.0594f
#define WAVES_PER_BLOCK 8
#define MT    2            // M-tiles (of 16 windows) per wave -> 32 windows/wave
#define SM_HALVES 576      // samples staged per wave: 31*4 + 442 = 566 -> 576
#define NCHUNK 512         // scan parallelization chunks

// ---------------------------------------------------------------------------
// Build the DFT matrix (cos / -sin), pre-swizzled into WMMA B-fragment layout.
// Fragment (nt, ks): 32 lanes x 16 halves; element i of lane l corresponds to
//   N = nt*16 + (l & 15),  K = ks*32 + (l>>4)*8 + (i<8 ? i : i+8)     (ISA 16-bit B layout)
// Out-of-range N (>=221) / K (>=442) are zero so padding is inert.
// ---------------------------------------------------------------------------
__global__ void build_dft(_Float16* __restrict__ bre, _Float16* __restrict__ bim) {
    int tid = blockIdx.x * blockDim.x + threadIdx.x;
    const int total = NNT * NKS * 32 * 16;
    if (tid >= total) return;
    int i    = tid & 15;
    int lane = (tid >> 4) & 31;
    int frag = tid >> 9;            // nt*NKS + ks
    int ks   = frag % NKS;
    int nt   = frag / NKS;
    int N = nt * 16 + (lane & 15);
    int K = ks * 32 + ((lane >> 4) << 3) + (i < 8 ? i : i + 8);
    float re = 0.0f, im = 0.0f;
    if (N < NBINS && K < WIN) {
        int r = (N * K) % WIN;      // exact periodic reduction
        float theta = 6.28318530717958647692f * (float)r / (float)WIN;
        re =  cosf(theta);
        im = -sinf(theta);          // fft convention exp(-i*2*pi*jk/N)
    }
    bre[tid] = (_Float16)re;
    bim[tid] = (_Float16)im;
}

// ---------------------------------------------------------------------------
// One wave per pair of M-tiles (32 windows). Windows m0..m0+31 cover samples
// [m0*4, m0*4 + 565] -> stage 576 contiguous samples in LDS as f16.
// For each of 14 N-tiles: accumulate Re and Im for both M-tiles over 14
// K-steps; each B fragment load feeds 4 WMMAs (32 FLOP per L2 byte).
// Then threshold + |Re|+|Im| + running argmax + cross-lane argmax.
// ---------------------------------------------------------------------------
__global__ __launch_bounds__(WAVES_PER_BLOCK * 32)
void dft_argmax(const float* __restrict__ x, int n, int nwin,
                const _Float16* __restrict__ bre, const _Float16* __restrict__ bim,
                float* __restrict__ tmp) {
    __shared__ _Float16 smem[WAVES_PER_BLOCK * SM_HALVES];

    int lane = threadIdx.x & 31;
    int wave = threadIdx.x >> 5;
    int pt   = blockIdx.x * WAVES_PER_BLOCK + wave;   // pair-tile index
    int npt  = (nwin + (16 * MT - 1)) / (16 * MT);
    if (pt >= npt) return;                  // wave-uniform: EXEC stays all-1s

    _Float16* sm = &smem[wave * SM_HALVES];
    int base = pt * (64 * MT);              // first sample of this pair-tile

    if (base + SM_HALVES <= n) {            // fast path: vectorized loads
        const float4* src = (const float4*)(x + base);
        #pragma unroll
        for (int r = 0; r < 4; ++r) {
            float4 v = src[lane + r * 32];
            int o = (lane + r * 32) << 2;
            sm[o + 0] = (_Float16)v.x;
            sm[o + 1] = (_Float16)v.y;
            sm[o + 2] = (_Float16)v.z;
            sm[o + 3] = (_Float16)v.w;
        }
        float2 v2 = ((const float2*)(x + base + 512))[lane];   // samples 512..575
        sm[512 + lane * 2 + 0] = (_Float16)v2.x;
        sm[512 + lane * 2 + 1] = (_Float16)v2.y;
    } else {                                // tail tile: clamped scalar loads
        for (int e = lane; e < SM_HALVES; e += 32) {
            int idx = base + e;
            if (idx >= n) idx = n - 1;
            sm[e] = (_Float16)x[idx];
        }
    }
    // LDS write -> read by same wave: DS ops are in-order; wait + compiler fence.
    asm volatile("s_wait_dscnt 0" ::: "memory");

    const unsigned* smu = (const unsigned*)sm;
    int m  = lane & 15;                     // A-matrix row (window in tile)
    int kb = (lane >> 4) << 3;              // K sub-base per lane group (0 or 8)

    float best[MT][8];
    int   bestn[MT][8];
    #pragma unroll
    for (int mt = 0; mt < MT; ++mt)
        #pragma unroll
        for (int v = 0; v < 8; ++v) { best[mt][v] = -1.0f; bestn[mt][v] = 0; }

    for (int nt = 0; nt < NNT; ++nt) {
        v8f cre[MT], cim[MT];
        #pragma unroll
        for (int mt = 0; mt < MT; ++mt) { cre[mt] = (v8f){}; cim[mt] = (v8f){}; }

        #pragma unroll
        for (int ks = 0; ks < NKS; ++ks) {
            int fo = (((nt * NKS + ks) << 5) + lane) << 4;   // fragment offset (halves)
            const v16h bR = *(const v16h*)(bre + fo);
            const v16h bI = *(const v16h*)(bim + fo);
            #pragma unroll
            for (int mt = 0; mt < MT; ++mt) {
                // A fragment: element i -> K = ks*32 + kb + (i<8 ? i : i+8);
                // pairs are K-consecutive & even -> gather 8 dwords from LDS.
                v16h a;
                unsigned* ap = (unsigned*)&a;
                #pragma unroll
                for (int p = 0; p < 8; ++p) {
                    int kp = ks * 32 + kb + (p < 4 ? 2 * p : 2 * p + 8);
                    ap[p] = smu[(mt * 64 + (m << 2) + kp) >> 1];  // sample mt*64 + m*4 + K
                }
                cre[mt] = __builtin_amdgcn_wmma_f32_16x16x32_f16(false, a, false, bR,
                                                                 (short)0, cre[mt], false, false);
                cim[mt] = __builtin_amdgcn_wmma_f32_16x16x32_f16(false, a, false, bI,
                                                                 (short)0, cim[mt], false, false);
            }
        }
        // C layout: element v of lane l -> window row M = v + (l>>4)*8, bin = l&15
        int nbin = (nt << 4) + (lane & 15);
        bool nvalid = nbin < NBINS;
        #pragma unroll
        for (int mt = 0; mt < MT; ++mt) {
            #pragma unroll
            for (int v = 0; v < 8; ++v) {
                float re = cre[mt][v], im = cim[mt][v];
                float mag = fabsf(re) + fabsf(im);
                if (re * re + im * im < THR2) mag = 0.0f;   // threshold |spec| < 0.01
                if (!nvalid) mag = -1.0f;                   // exclude padded bins
                if (mag > best[mt][v]) { best[mt][v] = mag; bestn[mt][v] = nbin; }
            }
        }
    }

    // Cross-lane argmax over the 16 lanes holding one window's bins.
    // Tie-break: lowest bin index (matches jnp.argmax first-occurrence).
    #pragma unroll
    for (int mt = 0; mt < MT; ++mt) {
        #pragma unroll
        for (int v = 0; v < 8; ++v) {
            float bv = best[mt][v];
            int   bn = bestn[mt][v];
            #pragma unroll
            for (int off = 1; off < 16; off <<= 1) {
                float ov = __shfl_xor(bv, off, 32);
                int   on = __shfl_xor(bn, off, 32);
                if (ov > bv || (ov == bv && on < bn)) { bv = ov; bn = on; }
            }
            if ((lane & 15) == 0) {
                int mrow = v + ((lane >> 4) << 3);
                int widx = pt * (16 * MT) + mt * 16 + mrow;
                if (widx < nwin) tmp[widx] = (float)bn * 100.0f;
            }
        }
    }
}

// ---------------------------------------------------------------------------
// Parallel exact scan, phase 1: per-chunk transfer-function table.
// maxF is always a multiple of 100 in [0, 22000] -> only 221 possible entry
// states. Each block stages its chunk in LDS; thread e simulates entry
// maxF = 100*e and records exit (maxF, Ftime, accepted-anything?).
// If nothing accepted, exit state is the (unknown) entry state -> passthrough.
// ---------------------------------------------------------------------------
__global__ void scan_table(const float* __restrict__ tmp, int nwin, int L,
                           float* __restrict__ exitF, float* __restrict__ exitT,
                           int* __restrict__ exitA) {
    extern __shared__ float sch[];
    int c  = blockIdx.x;
    int b0 = c * L;
    int len = nwin - b0;
    if (len > L) len = L;
    if (len < 0) len = 0;
    for (int j = threadIdx.x; j < len; j += blockDim.x) sch[j] = tmp[b0 + j];
    __syncthreads();

    int e = threadIdx.x;
    if (e >= NBINS) return;
    float maxF = 100.0f * (float)e;
    float Ft   = 0.0f;
    int   acc  = 0;
    for (int j = 0; j < len; ++j) {
        float t = sch[j];
        if (fabsf(maxF - t) > maxF * RELC) {
            maxF = t;
            Ft   = (float)((b0 + j) * HOP);
            acc  = 1;
        }
    }
    int o = c * NBINS + e;
    exitF[o] = maxF;
    exitT[o] = Ft;
    exitA[o] = acc;
}

// ---------------------------------------------------------------------------
// Parallel exact scan, phase 2: chain chunk transfer functions (NCHUNK steps,
// single thread) to get each chunk's true entry state; also emit `final`.
// ---------------------------------------------------------------------------
__global__ void scan_link(const float* __restrict__ exitF, const float* __restrict__ exitT,
                          const int* __restrict__ exitA, int nchunk, int nwin,
                          float* __restrict__ entryF, float* __restrict__ entryT,
                          float* __restrict__ out) {
    if (blockIdx.x != 0 || threadIdx.x != 0) return;
    float maxF = 0.0f, Ft = 0.0f;
    for (int c = 0; c < nchunk; ++c) {
        entryF[c] = maxF;
        entryT[c] = Ft;
        int k = (int)(maxF * 0.01f + 0.5f);     // maxF == 100*k exactly
        int o = c * NBINS + k;
        if (exitA[o]) { maxF = exitF[o]; Ft = exitT[o]; }
        // else: passthrough, state unchanged
    }
    const float ms = 1000.0f / 44100.0f;
    float lastI = (float)((nwin - 1) * HOP);
    out[3 * nwin + 0] = maxF;
    out[3 * nwin + 1] = (lastI - Ft) * ms;
}

// ---------------------------------------------------------------------------
// Parallel exact scan, phase 3: each thread replays one chunk with its true
// entry state and writes flags/freqs/durs. Bit-identical to sequential scan.
// out layout: [flags(nwin) | freqs(nwin) | durs(nwin) | final(2)]
// ---------------------------------------------------------------------------
__global__ void scan_emit(const float* __restrict__ tmp, int nwin, int L, int nchunk,
                          const float* __restrict__ entryF, const float* __restrict__ entryT,
                          float* __restrict__ out) {
    int c = blockIdx.x * blockDim.x + threadIdx.x;
    if (c >= nchunk) return;
    float maxF = entryF[c], Ft = entryT[c];
    float* flags = out;
    float* freqs = out + nwin;
    float* durs  = out + 2 * nwin;
    const float ms = 1000.0f / 44100.0f;
    int b0  = c * L;
    int end = b0 + L;
    if (end > nwin) end = nwin;
    for (int i = b0; i < end; ++i) {
        float t  = tmp[i];
        float is = (float)(i * HOP);
        bool cond = fabsf(maxF - t) > maxF * RELC;
        flags[i] = cond ? 1.0f : 0.0f;
        freqs[i] = maxF;
        durs[i]  = (is - Ft) * ms;
        if (cond) { maxF = t; Ft = is; }
    }
}

// ---------------------------------------------------------------------------
extern "C" void kernel_launch(void* const* d_in, const int* in_sizes, int n_in,
                              void* d_out, int out_size, void* d_ws, size_t ws_size,
                              hipStream_t stream) {
    (void)n_in; (void)out_size; (void)ws_size;
    const float* x = (const float*)d_in[0];
    int n    = in_sizes[0];
    int nwin = (n - WIN + HOP - 1) / HOP;

    // workspace layout (each region 256B-aligned):
    //   [tmp: nwin f32][B_re][B_im][exitF][exitT][exitA][entryF][entryT]
    const int bhalves = NNT * NKS * 32 * 16;          // 100,352 halves per matrix
    char* ws = (char*)d_ws;
    size_t off = 0;
    auto take = [&](size_t bytes) { char* p = ws + off; off = (off + bytes + 255) & ~(size_t)255; return p; };
    float*    tmp    = (float*)   take((size_t)nwin * sizeof(float));
    _Float16* bre    = (_Float16*)take((size_t)bhalves * sizeof(_Float16));
    _Float16* bim    = (_Float16*)take((size_t)bhalves * sizeof(_Float16));
    float*    exitF  = (float*)   take((size_t)NCHUNK * NBINS * sizeof(float));
    float*    exitT  = (float*)   take((size_t)NCHUNK * NBINS * sizeof(float));
    int*      exitA  = (int*)     take((size_t)NCHUNK * NBINS * sizeof(int));
    float*    entryF = (float*)   take((size_t)NCHUNK * sizeof(float));
    float*    entryT = (float*)   take((size_t)NCHUNK * sizeof(float));

    build_dft<<<(bhalves + 255) / 256, 256, 0, stream>>>(bre, bim);

    int npt    = (nwin + (16 * MT - 1)) / (16 * MT);
    int blocks = (npt + WAVES_PER_BLOCK - 1) / WAVES_PER_BLOCK;
    dft_argmax<<<blocks, WAVES_PER_BLOCK * 32, 0, stream>>>(x, n, nwin, bre, bim, tmp);

    int L = (nwin + NCHUNK - 1) / NCHUNK;             // chunk length (~489)
    scan_table<<<NCHUNK, 224, (size_t)L * sizeof(float), stream>>>(tmp, nwin, L,
                                                                   exitF, exitT, exitA);
    scan_link<<<1, 32, 0, stream>>>(exitF, exitT, exitA, NCHUNK, nwin,
                                    entryF, entryT, (float*)d_out);
    scan_emit<<<(NCHUNK + 255) / 256, 256, 0, stream>>>(tmp, nwin, L, NCHUNK,
                                                        entryF, entryT, (float*)d_out);
}
